// GA_44495861186995
// MI455X (gfx1250) — compile-verified, matrix operand order checked by reading, not run on hardware
//
#include <hip/hip_runtime.h>
#include <math.h>
#include <stdint.h>

// ---------------------------------------------------------------------------
// Problem constants (match reference)
// ---------------------------------------------------------------------------
#define B_    128
#define C_    2048
#define L_    196
#define LP2_  264          // storage stride for padded-L rows (zero-filled tail,
                           // 264*2B = 528B = 16B-aligned rows; covers l0+63+tap)
#define K3_   6144         // im2col K = 3 * C
#define H_    1024         // MLP hidden = C/2
#define DATT_ 312
#define NCLS_ 200
#define TOPK_ 120

typedef __attribute__((ext_vector_type(16))) __bf16 v16bf;
typedef __attribute__((ext_vector_type(8)))  float  v8f;

union U16x8 { uint4 v; __bf16 h[8]; };

// ---------------------------------------------------------------------------
// CDNA5 async copy global->LDS (ASYNCcnt) — 16B per lane.
// CDNA5 ISA: GLOBAL_LOAD_ASYNC_TO_LDS_B128, S_WAIT_ASYNCCNT.
// LDS offset = low 32 bits of the generic pointer (aperture in high bits).
// ---------------------------------------------------------------------------
static __device__ __forceinline__ void async_copy_b128(void* lds_ptr, const void* g) {
  unsigned lds_off = (unsigned)(uintptr_t)lds_ptr;
  asm volatile("global_load_async_to_lds_b128 %0, %1, off"
               :: "v"(lds_off), "v"((unsigned long long)(uintptr_t)g)
               : "memory");
}
static __device__ __forceinline__ void wait_async() {
  asm volatile("s_wait_asynccnt 0x0" ::: "memory");
}

// ---------------------------------------------------------------------------
// WMMA helpers (gfx1250: V_WMMA_F32_16X16X32_BF16, wave32)
// Fragment layout per CDNA5 ISA 7.12.2:
//   lane = hi*16 + m ; elements 0..7 -> K = hi*8 + i ; 8..15 -> K = 16 + hi*8 + i
//   (same per-lane pattern for B with N across lanes; LDS tiles are stored
//    row-per-M (or per-N) with K contiguous)
//   C/D v8f: element e -> M = e + 8*(lane>=16), N = lane&15
// ---------------------------------------------------------------------------
static __device__ __forceinline__ v16bf load_frag(const __bf16* base, int stride, int lane) {
  const int m  = lane & 15;
  const int hi = (lane >> 4) & 1;
  const __bf16* p = base + m * stride + hi * 8;
  v16bf r;
#pragma unroll
  for (int i = 0; i < 8; ++i) { r[i] = p[i]; r[i + 8] = p[16 + i]; }
  return r;
}

static __device__ __forceinline__ v8f wmma_bf16(v16bf a, v16bf b, v8f c) {
  return __builtin_amdgcn_wmma_f32_16x16x32_bf16(false, a, false, b, (short)0, c,
                                                 false, false);
}

static __device__ __forceinline__ float sigf(float x) { return 1.0f / (1.0f + expf(-x)); }

// ---------------------------------------------------------------------------
// Prep kernels
// ---------------------------------------------------------------------------

// convQ_w (C,C,3) fp32 -> Wbf[co][tap*2048+ci] bf16 (row-major, lda = 6144)
__global__ void k_prep_w(const float* __restrict__ w, __bf16* __restrict__ Wbf) {
  size_t i = (size_t)blockIdx.x * 256 + threadIdx.x;
  if (i >= (size_t)C_ * K3_) return;
  int co = (int)(i / K3_);
  int r  = (int)(i % K3_);
  int tap = r >> 11, ci = r & 2047;
  Wbf[i] = (__bf16)w[((size_t)co * C_ + ci) * 3 + tap];
}

// Fs (B,197,C) -> Inpad (B,C,LP2_) bf16: col 0 = 0, cols 1..196 = data,
// cols 197..263 = 0.  LDS tile transpose for coalescing both sides.
__global__ __launch_bounds__(256) void k_prep_in(const float* __restrict__ Fs,
                                                 __bf16* __restrict__ Inpad) {
  __shared__ float T[32 * 65];
  const int b = blockIdx.z, ci0 = blockIdx.x * 32, l0 = blockIdx.y * 64;
  const int t = threadIdx.x;
  {
    int ci = t & 31, lr = t >> 5;
#pragma unroll
    for (int j = 0; j < 8; ++j) {
      int l = l0 + lr * 8 + j;
      float v = 0.f;
      if (l < L_) v = Fs[((size_t)b * 197 + (l + 1)) * C_ + ci0 + ci];
      T[ci * 65 + lr * 8 + j] = v;
    }
  }
  __syncthreads();
  {
    int cio = t >> 3, seg = (t & 7) * 8;
    size_t rb = ((size_t)b * C_ + ci0 + cio) * LP2_;
#pragma unroll
    for (int i = 0; i < 8; ++i) {
      int l = l0 + seg + i;
      if (l < L_) Inpad[rb + 1 + l] = (__bf16)T[cio * 65 + seg + i];
    }
    if (l0 == 0 && t < 32)
      Inpad[((size_t)b * C_ + ci0 + t) * LP2_] = (__bf16)0.f;
    if (l0 == 192) {  // zero-fill tail cols 197..263
      for (int idx = t; idx < 32 * 67; idx += 256) {
        int ci = idx / 67, col = 197 + idx % 67;
        Inpad[((size_t)b * C_ + ci0 + ci) * LP2_ + col] = (__bf16)0.f;
      }
    }
  }
}

// Fg[b,c] = Fs[b,0,c]
__global__ void k_fg(const float* __restrict__ Fs, float* __restrict__ Fg) {
  int i = blockIdx.x * 256 + threadIdx.x;
  int b = i >> 11, c = i & 2047;
  Fg[i] = Fs[(size_t)b * 197 * C_ + c];
}

// L2 norms of w_q rows
__global__ __launch_bounds__(256) void k_wqnorm(const float* __restrict__ wq,
                                                float* __restrict__ norms) {
  __shared__ float red[256];
  const int v = blockIdx.x, t = threadIdx.x;
  const float* r = wq + (size_t)v * C_;
  float s = 0.f;
  for (int i = t; i < C_; i += 256) { float x = r[i]; s += x * x; }
  red[t] = s; __syncthreads();
  for (int off = 128; off > 0; off >>= 1) { if (t < off) red[t] += red[t + off]; __syncthreads(); }
  if (t == 0) norms[v] = fmaxf(sqrtf(red[0]), 1e-12f);
}

// wqn[v][f] = bf16(wq[v,f]/norm[v]) row-major, 256 rows (rows 196..255 zero)
__global__ void k_wqn(const float* __restrict__ wq, const float* __restrict__ norms,
                      __bf16* __restrict__ wqn) {
  size_t i = (size_t)blockIdx.x * 256 + threadIdx.x;
  if (i >= (size_t)256 * C_) return;
  int v = (int)(i >> 11);
  __bf16 val = (__bf16)0.f;
  if (v < L_) val = (__bf16)(wq[i] / norms[v]);
  wqn[i] = val;
}

// ---------------------------------------------------------------------------
// Conv-as-GEMM: Ft[b,co,l] = sigmoid( sum_kk Wbf[co,kk]*B[kk,l] + bias[co] )
// B[kk,l] = Inpad[b, kk%2048, l + kk/2048]   (im2col, 3 taps, zero-padded rows)
// Block tile 128x64, K-step 32, 8 waves 4x2, wave tile 32x32.
// Double-buffered LDS; A tiles via GLOBAL_LOAD_ASYNC_TO_LDS_B128.
// ---------------------------------------------------------------------------
#define ASTR 40  // 80B rows: 16B-aligned for async b128; 20-bank advance, conflict-free
#define BSTR 34  // 17-bank advance, conflict-free

__global__ __launch_bounds__(256) void k_conv(const __bf16* __restrict__ Wbf,
                                              const __bf16* __restrict__ Inpad,
                                              const float* __restrict__ bias,
                                              float* __restrict__ Ft) {
  __shared__ __bf16 As[2][128 * ASTR];
  __shared__ __bf16 Bs[2][64 * BSTR];
  const int tid = threadIdx.x;
  const int b = blockIdx.z;
  const int co0 = blockIdx.x * 128;
  const int l0  = blockIdx.y * 64;
  const int lane = tid & 31, wave = tid >> 5;
  const int wm = wave & 3, wn = wave >> 2;

  // per-thread copy coordinates
  const int arow = tid >> 2, aq = (tid & 3) * 8;          // A: 2 chunks of 8 bf16 (16B)
  const int bkr = tid >> 3, bseg = (tid & 7) * 8;         // B: 8 bf16, im2col

  v8f acc[4];
#pragma unroll
  for (int i = 0; i < 4; ++i)
#pragma unroll
    for (int e = 0; e < 8; ++e) acc[i][e] = 0.f;

  __bf16 breg[8];

  // ---- prologue: tile 0 into buffer 0
  {
    const int kk0 = 0;
    async_copy_b128(&As[0][arow * ASTR + aq],
                    Wbf + (size_t)(co0 + arow) * K3_ + kk0 + aq);
    async_copy_b128(&As[0][(arow + 64) * ASTR + aq],
                    Wbf + (size_t)(co0 + arow + 64) * K3_ + kk0 + aq);
    const __bf16* gp = Inpad + ((size_t)b * C_ + bkr) * LP2_ + l0 + bseg;  // tap 0
#pragma unroll
    for (int i = 0; i < 8; ++i) breg[i] = gp[i];
#pragma unroll
    for (int i = 0; i < 8; ++i) Bs[0][(bseg + i) * BSTR + bkr] = breg[i];
  }
  wait_async();
  __syncthreads();

  for (int it = 0; it < 192; ++it) {
    const int cur = it & 1, nxt = cur ^ 1;
    const bool more = (it < 191);
    if (more) {  // issue next A tile (async) + next B tile global loads
      const int kk0 = (it + 1) * 32;
      async_copy_b128(&As[nxt][arow * ASTR + aq],
                      Wbf + (size_t)(co0 + arow) * K3_ + kk0 + aq);
      async_copy_b128(&As[nxt][(arow + 64) * ASTR + aq],
                      Wbf + (size_t)(co0 + arow + 64) * K3_ + kk0 + aq);
      const int tap = kk0 >> 11, ci0 = kk0 & 2047;   // tile never straddles a tap
      const __bf16* gp = Inpad + ((size_t)b * C_ + ci0 + bkr) * LP2_ + l0 + bseg + tap;
#pragma unroll
      for (int i = 0; i < 8; ++i) breg[i] = gp[i];
    }

    // compute on current buffer
    v16bf a0 = load_frag(As[cur] + (wm * 32 + 0)  * ASTR, ASTR, lane);
    v16bf a1 = load_frag(As[cur] + (wm * 32 + 16) * ASTR, ASTR, lane);
    v16bf b0 = load_frag(Bs[cur] + (wn * 32 + 0)  * BSTR, BSTR, lane);
    v16bf b1 = load_frag(Bs[cur] + (wn * 32 + 16) * BSTR, BSTR, lane);
    acc[0] = wmma_bf16(a0, b0, acc[0]);
    acc[1] = wmma_bf16(a0, b1, acc[1]);
    acc[2] = wmma_bf16(a1, b0, acc[2]);
    acc[3] = wmma_bf16(a1, b1, acc[3]);

    if (more) {
#pragma unroll
      for (int i = 0; i < 8; ++i) Bs[nxt][(bseg + i) * BSTR + bkr] = breg[i];
      wait_async();
    }
    __syncthreads();
  }

  const int hi = lane >> 4, n = lane & 15;
#pragma unroll
  for (int fi = 0; fi < 2; ++fi)
#pragma unroll
    for (int fj = 0; fj < 2; ++fj) {
      v8f c = acc[fi * 2 + fj];
#pragma unroll
      for (int e = 0; e < 8; ++e) {
        int m  = e + hi * 8;
        int co = co0 + wm * 32 + fi * 16 + m;
        int l  = l0  + wn * 32 + fj * 16 + n;
        if (l < L_) {
          float v = c[e] + bias[co];
          Ft[((size_t)b * C_ + co) * L_ + l] = sigf(v);
        }
      }
    }
}

// ---------------------------------------------------------------------------
// G[b,f,l] = bf16(Ft[b,f,l]*Fg[b,f]) into reused Inpad buffer (stride LP2_,
// tail cols zeroed) + ftmax.  One wave per row.
// ---------------------------------------------------------------------------
__global__ __launch_bounds__(256) void k_gform(const float* __restrict__ Ft,
                                               const float* __restrict__ Fg,
                                               __bf16* __restrict__ G,
                                               float* __restrict__ ftmax) {
  const int lane = threadIdx.x & 31, wid = threadIdx.x >> 5;
  const size_t row = (size_t)blockIdx.x * 8 + wid;
  const float* fr = Ft + row * L_;
  const float fg = Fg[row];
  __bf16* gr = G + row * LP2_;
  float m = -1.f;
#pragma unroll
  for (int j = 0; j < 7; ++j) {
    int l = j * 32 + lane;
    if (l < L_) {
      float v = fr[l];
      m = fmaxf(m, v);
      gr[l] = (__bf16)(v * fg);
    }
  }
  for (int l = L_ + lane; l < LP2_; l += 32) gr[l] = (__bf16)0.f;  // zero tail
  for (int off = 16; off > 0; off >>= 1) m = fmaxf(m, __shfl_xor(m, off, 32));
  if (lane == 0) ftmax[row] = m;
}

// ---------------------------------------------------------------------------
// scores GEMM: scores[b,l,v] = sum_f G[b,f,l] * wqn[v,f]
// Block tile 64x64, K-step 32, 8 waves 2x4, wave tile 32x16.
// A: LDS-transposed from f-major G (aligned uint4 reads, unguarded);
// B: async b128 straight copies from wqn rows.
// ---------------------------------------------------------------------------
__global__ __launch_bounds__(256) void k_scores(const __bf16* __restrict__ G,
                                                const __bf16* __restrict__ wqn,
                                                float* __restrict__ scores) {
  __shared__ __bf16 As2[64 * BSTR];
  __shared__ __bf16 Bs2[64 * ASTR];
  const int tid = threadIdx.x;
  const int b = blockIdx.z;
  const int l0 = blockIdx.x * 64;
  const int v0 = blockIdx.y * 64;
  const int lane = tid & 31, wave = tid >> 5;
  const int wm = wave & 1, wn = wave >> 1;

  v8f acc[2];
#pragma unroll
  for (int i = 0; i < 2; ++i)
#pragma unroll
    for (int e = 0; e < 8; ++e) acc[i][e] = 0.f;

  for (int f0 = 0; f0 < C_; f0 += 32) {
    __syncthreads();
    {  // B tile: rows n=v, K-contiguous -> direct async copies
      int row = tid >> 2, q = (tid & 3) * 8;
      async_copy_b128(&Bs2[row * ASTR + q],
                      wqn + (size_t)(v0 + row) * C_ + f0 + q);
    }
    {  // A tile: rows m=l, cols k=f ; G is f-major -> transpose via LDS
      int k = tid >> 3, seg = (tid & 7) * 8;
      const __bf16* gp = G + ((size_t)b * C_ + f0 + k) * LP2_ + l0 + seg;
      U16x8 u; u.v = *(const uint4*)gp;   // 16B-aligned, zero-padded tail
#pragma unroll
      for (int i = 0; i < 8; ++i) As2[(seg + i) * BSTR + k] = u.h[i];
    }
    wait_async();
    __syncthreads();
    v16bf a0 = load_frag(As2 + (wm * 32 + 0)  * BSTR, BSTR, lane);
    v16bf a1 = load_frag(As2 + (wm * 32 + 16) * BSTR, BSTR, lane);
    v16bf bb = load_frag(Bs2 + (wn * 16)      * ASTR, ASTR, lane);
    acc[0] = wmma_bf16(a0, bb, acc[0]);
    acc[1] = wmma_bf16(a1, bb, acc[1]);
  }

  const int hi = lane >> 4, n = lane & 15;
#pragma unroll
  for (int fi = 0; fi < 2; ++fi) {
    v8f c = acc[fi];
#pragma unroll
    for (int e = 0; e < 8; ++e) {
      int m = e + hi * 8;
      int l = l0 + wm * 32 + fi * 16 + m;
      int v = v0 + wn * 16 + n;
      if (l < L_ && v < L_) scores[((size_t)b * L_ + l) * L_ + v] = c[e];
    }
  }
}

// ---------------------------------------------------------------------------
// Per batch: softmax over v, mean over l, l2-norm, CCL mask (14x14, 196 iters,
// area>=30), sA[l] = sigmoid(A_masked[l]).  One block per batch.
// ---------------------------------------------------------------------------
__global__ __launch_bounds__(256) void k_softmax_ccl(const float* __restrict__ scores,
                                                     float* __restrict__ sA) {
  __shared__ float rm[L_], rs[L_], Av[L_], red[256];
  __shared__ int lab[L_], cnt[L_ + 1];
  const int b = blockIdx.x, t = threadIdx.x;
  const float* sc = scores + (size_t)b * L_ * L_;

  if (t < L_) {
    float m = -3.4e38f;
    for (int v = 0; v < L_; ++v) m = fmaxf(m, sc[t * L_ + v]);
    float s = 0.f;
    for (int v = 0; v < L_; ++v) s += expf(sc[t * L_ + v] - m);
    rm[t] = m; rs[t] = s;
  }
  __syncthreads();
  if (t < L_) {
    float a = 0.f;
    for (int l = 0; l < L_; ++l) a += expf(sc[l * L_ + t] - rm[l]) / rs[l];
    Av[t] = a * (1.0f / L_);
  }
  __syncthreads();
  red[t] = (t < L_) ? Av[t] * Av[t] : 0.f;
  __syncthreads();
  for (int off = 128; off > 0; off >>= 1) { if (t < off) red[t] += red[t + off]; __syncthreads(); }
  float scale = 1.0f / fmaxf(sqrtf(red[0]), 1e-12f);
  __syncthreads();
  if (t < L_) Av[t] *= scale;
  if (t <= L_) cnt[t] = 0;
  __syncthreads();

  bool bin = (t < L_) && (Av[t] > 0.05f);
  if (t < L_) lab[t] = bin ? t : L_;
  __syncthreads();

  for (int it = 0; it < L_; ++it) {
    int nl = L_;
    if (t < L_ && bin) {
      int r = t / 14, c = t % 14;
      int mn = lab[t];
      for (int dr = -1; dr <= 1; ++dr)
        for (int dc = -1; dc <= 1; ++dc) {
          int rr = r + dr, cc = c + dc;
          if (rr >= 0 && rr < 14 && cc >= 0 && cc < 14) mn = min(mn, lab[rr * 14 + cc]);
        }
      nl = mn;
    }
    __syncthreads();
    if (t < L_) lab[t] = nl;
    __syncthreads();
  }

  if (t < L_ && bin) atomicAdd(&cnt[lab[t]], 1);
  __syncthreads();
  if (t < L_) {
    bool filt = bin && (cnt[lab[t]] >= 30);
    float a = filt ? Av[t] : 0.f;
    sA[(size_t)b * L_ + t] = sigf(a);
  }
}

// ---------------------------------------------------------------------------
// One wave per (b,c) row: Fk = Ft*sA, fkmax = max(Fk), exact top-120 mask
// (ties -> earliest index like lax.top_k) written back in place.
// Threshold found by 31-step binary search over float bit patterns.
// ---------------------------------------------------------------------------
__global__ __launch_bounds__(256) void k_fk_topk(float* __restrict__ FtOut,
                                                 const float* __restrict__ sA,
                                                 float* __restrict__ fkmax) {
  const int lane = threadIdx.x & 31, wid = threadIdx.x >> 5;
  const size_t row = (size_t)blockIdx.x * 8 + wid;
  const int b = (int)(row >> 11);
  float* fr = FtOut + row * L_;
  const float* sa = sA + (size_t)b * L_;

  float v[7];
  float m = -1.f;
#pragma unroll
  for (int j = 0; j < 7; ++j) {
    int l = j * 32 + lane;
    if (l < L_) { v[j] = fr[l] * sa[l]; m = fmaxf(m, v[j]); }
    else v[j] = -1.f;  // sentinel; real values are > 0
  }
  for (int off = 16; off > 0; off >>= 1) m = fmaxf(m, __shfl_xor(m, off, 32));
  if (lane == 0) fkmax[row] = m;

  unsigned lo = 0u, hi = 0x7f7fffffu;
  while (lo < hi) {
    unsigned mid = lo + (hi - lo + 1u) / 2u;
    float tf = __uint_as_float(mid);
    int cnt = 0;
#pragma unroll
    for (int j = 0; j < 7; ++j) cnt += __popcll(__ballot(v[j] >= tf));
    if (cnt >= TOPK_) lo = mid; else hi = mid - 1u;
  }
  const float T = __uint_as_float(lo);

  int g = 0;
  unsigned eqb[7];
#pragma unroll
  for (int j = 0; j < 7; ++j) {
    g += __popcll(__ballot(v[j] > T));
    eqb[j] = (unsigned)__ballot(v[j] == T);
  }
  const int r = TOPK_ - g;
  const unsigned lmask = (1u << lane) - 1u;
  int pre = 0;
#pragma unroll
  for (int j = 0; j < 7; ++j) {
    int l = j * 32 + lane;
    if (l < L_) {
      int rank = pre + __popc(eqb[j] & lmask);
      bool keep = (v[j] > T) || (v[j] == T && rank < r);
      fr[l] = keep ? v[j] : 0.f;
    }
    pre += __popc(eqb[j]);
  }
}

// ---------------------------------------------------------------------------
// Small fp32 FC layer batched over M=128: Y[128,N] = act(X @ W^T + b)
// ---------------------------------------------------------------------------
__global__ __launch_bounds__(256) void k_fc(const float* __restrict__ X,
                                            const float* __restrict__ W,
                                            const float* __restrict__ bias,
                                            float* __restrict__ Y,
                                            int N, int K, int relu) {
  __shared__ float Xs[128 * 33];
  __shared__ float Ws[64 * 33];
  const int t = threadIdx.x;
  const int n0 = blockIdx.x * 64;
  const int ty = t >> 4, tx = t & 15;
  float acc[8][4];
#pragma unroll
  for (int i = 0; i < 8; ++i)
#pragma unroll
    for (int j = 0; j < 4; ++j) acc[i][j] = 0.f;

  for (int k0 = 0; k0 < K; k0 += 32) {
    __syncthreads();
    { int mm = t >> 1, seg = (t & 1) * 16;
      const float* gp = X + (size_t)mm * K + k0 + seg;
#pragma unroll
      for (int i = 0; i < 16; ++i) Xs[mm * 33 + seg + i] = gp[i]; }
    { int n = t >> 2, seg = (t & 3) * 8;
      int rown = n0 + n;
#pragma unroll
      for (int i = 0; i < 8; ++i) {
        float vv = 0.f;
        if (rown < N) vv = W[(size_t)rown * K + k0 + seg + i];
        Ws[n * 33 + seg + i] = vv; } }
    __syncthreads();
    for (int kk = 0; kk < 32; ++kk) {
      float a[8], w[4];
#pragma unroll
      for (int i = 0; i < 8; ++i) a[i] = Xs[(ty * 8 + i) * 33 + kk];
#pragma unroll
      for (int j = 0; j < 4; ++j) w[j] = Ws[(tx * 4 + j) * 33 + kk];
#pragma unroll
      for (int i = 0; i < 8; ++i)
#pragma unroll
        for (int j = 0; j < 4; ++j) acc[i][j] += a[i] * w[j];
    }
  }
#pragma unroll
  for (int i = 0; i < 8; ++i) {
    int mm = ty * 8 + i;
#pragma unroll
    for (int j = 0; j < 4; ++j) {
      int n = n0 + tx * 4 + j;
      if (n < N) {
        float vv = acc[i][j] + bias[n];
        if (relu) vv = fmaxf(vv, 0.f);
        Y[(size_t)mm * N + n] = vv;
      }
    }
  }
}

// ---------------------------------------------------------------------------
// Host-side orchestration
// ---------------------------------------------------------------------------
extern "C" void kernel_launch(void* const* d_in, const int* in_sizes, int n_in,
                              void* d_out, int out_size, void* d_ws, size_t ws_size,
                              hipStream_t stream) {
  (void)in_sizes; (void)n_in; (void)out_size; (void)ws_size;
  const float* Fs   = (const float*)d_in[0];
  const float* cw   = (const float*)d_in[1];
  const float* cb   = (const float*)d_in[2];
  const float* akw1 = (const float*)d_in[3];
  const float* akb1 = (const float*)d_in[4];
  const float* akw2 = (const float*)d_in[5];
  const float* akb2 = (const float*)d_in[6];
  const float* akw3 = (const float*)d_in[7];
  const float* akb3 = (const float*)d_in[8];
  const float* pw1  = (const float*)d_in[9];
  const float* pb1  = (const float*)d_in[10];
  const float* pw2  = (const float*)d_in[11];
  const float* pb2  = (const float*)d_in[12];
  const float* pw3  = (const float*)d_in[13];
  const float* pb3  = (const float*)d_in[14];
  const float* wq   = (const float*)d_in[15];

  float* out  = (float*)d_out;
  float* Ft   = out;                                   // (B,C,L); overwritten in place with Fs_out
  float* attO = out + (size_t)B_ * C_ * L_;            // (B,312)
  float* labO = attO + (size_t)B_ * DATT_;             // (B,200)

  char* ws = (char*)d_ws;
  size_t off = 0;
  auto alloc = [&](size_t bytes) { void* p = ws + off; off = (off + bytes + 255) & ~(size_t)255; return p; };
  __bf16* Wbf   = (__bf16*)alloc((size_t)C_ * K3_ * 2);        // 25 MB
  __bf16* Inpad = (__bf16*)alloc((size_t)B_ * C_ * LP2_ * 2);  // 138 MB (reused as G)
  __bf16* wqn   = (__bf16*)alloc((size_t)256 * C_ * 2);        // 1 MB
  float*  norms = (float*)alloc((size_t)L_ * 4);
  float*  Fg    = (float*)alloc((size_t)B_ * C_ * 4);
  float*  ftmax = (float*)alloc((size_t)B_ * C_ * 4);
  float*  fkmax = (float*)alloc((size_t)B_ * C_ * 4);
  float*  sc    = (float*)alloc((size_t)B_ * L_ * L_ * 4);     // 19.7 MB
  float*  sA    = (float*)alloc((size_t)B_ * L_ * 4);
  float*  h1    = (float*)alloc((size_t)B_ * H_ * 4);
  float*  h2    = (float*)alloc((size_t)B_ * H_ * 4);

  // 1) precision/layout prep
  k_prep_w <<<(int)(((size_t)C_ * K3_ + 255) / 256), 256, 0, stream>>>(cw, Wbf);
  k_prep_in<<<dim3(C_ / 32, 4, B_), 256, 0, stream>>>(Fs, Inpad);
  k_fg     <<<(B_ * C_) / 256, 256, 0, stream>>>(Fs, Fg);
  k_wqnorm <<<L_, 256, 0, stream>>>(wq, norms);
  k_wqn    <<<(256 * C_) / 256, 256, 0, stream>>>(wq, norms, wqn);

  // 2) conv1d as bf16 WMMA GEMM (dominant: ~631 GFLOP), async-LDS + dbl-buffer
  k_conv   <<<dim3(C_ / 128, 4, B_), 256, 0, stream>>>(Wbf, Inpad, cb, Ft);

  // 3) G = Ft*Fg (bf16, reuse Inpad) + ftmax
  k_gform  <<<(B_ * C_) / 8, 256, 0, stream>>>(Ft, Fg, Inpad, ftmax);

  // 4) scores GEMM (bf16 WMMA, ~20 GFLOP)
  k_scores <<<dim3(4, 4, B_), 256, 0, stream>>>(Inpad, wqn, sc);

  // 5) softmax/mean/l2norm/CCL -> sA = sigmoid(A_masked)
  k_softmax_ccl<<<B_, 256, 0, stream>>>(sc, sA);

  // 6) att MLP
  k_fc<<<H_ / 64, 256, 0, stream>>>(ftmax, akw1, akb1, h1, H_, C_, 1);
  k_fc<<<H_ / 64, 256, 0, stream>>>(h1, akw2, akb2, h2, H_, H_, 1);
  k_fc<<<(DATT_ + 63) / 64, 256, 0, stream>>>(h2, akw3, akb3, attO, DATT_, H_, 0);

  // 7) Fk = Ft*sA, exact top-120 mask in place, fkmax for classifier
  k_fk_topk<<<(B_ * C_) / 8, 256, 0, stream>>>(Ft, sA, fkmax);

  // 8) lable MLP
  k_fc<<<H_ / 64, 256, 0, stream>>>(fkmax, pw1, pb1, h1, H_, C_, 1);
  k_fc<<<H_ / 64, 256, 0, stream>>>(h1, pw2, pb2, h2, H_, H_, 1);
  k_fc<<<(NCLS_ + 63) / 64, 256, 0, stream>>>(h2, pw3, pb3, labO, NCLS_, H_, 0);
}
